// Decoder_2869038153809
// MI455X (gfx1250) — matrix operand order repaired and checked by reading, not compiled, not run on hardware
//
#include <hip/hip_runtime.h>
#include <hip/hip_bf16.h>
#include <stdint.h>

// ---------------- problem constants ----------------
#define NLAYER 6
#define BB     64
#define LL     52
#define VV     32000
#define DD     512
#define HHH    8
#define DKK    64
#define NPIXX  196
#define ENCC   2048
#define FFF    2048
#define MTOK   (BB * LL)       // 3328
#define MENC   (BB * NPIXX)    // 12544

// ---------------- vector types ----------------
typedef __bf16        bf16x16 __attribute__((ext_vector_type(16)));
typedef float         f32x8   __attribute__((ext_vector_type(8)));
typedef unsigned int  u32x4   __attribute__((ext_vector_type(4)));
typedef float         f32x4   __attribute__((ext_vector_type(4)));

__device__ __forceinline__ unsigned short f2bf(float f) {
  __bf16 h = (__bf16)f;                       // RNE hardware convert
  return __builtin_bit_cast(unsigned short, h);
}
__device__ __forceinline__ float bf2f(unsigned int lo16) {
  return __uint_as_float(lo16 << 16);
}

// CDNA5 async global->LDS copy (ASYNCcnt-tracked), GVS mode:
// mem = SGPR64 base + VGPR32 byte offset ; LDS byte addr in vdst VGPR.
__device__ __forceinline__ void async_cp_b128(unsigned lds_addr, unsigned goff,
                                              unsigned long long sbase) {
  asm volatile("global_load_async_to_lds_b128 %0, %1, %2"
               :: "v"(lds_addr), "v"(goff), "s"(sbase) : "memory");
}
__device__ __forceinline__ void wait_async0() {
  asm volatile("s_wait_asynccnt 0" ::: "memory");
}
__device__ __forceinline__ unsigned lds_off(const void* p) {
  return (unsigned)(unsigned long long)p;   // flat LDS aperture: offset = addr[31:0]
}

// =====================================================================
// WMMA GEMM:  C[M,N] = A[M,K] * Bt^T + bias (+ReLU) (+bf16 output)
//   A : bf16 [M,K] row-major (k-contiguous)
//   Bt: bf16 [N,K] row-major (k-contiguous, weights pre-transposed)
// Block tile 128x128x32, 128 threads = 4 waves (2M x 2N); each wave owns
// a 64x64 macro-tile = 4x4 WMMA tiles -> 16 v_wmma_f32_16x16x32_bf16 per
// k-tile from 4 A + 4 B fragments (wmma:ds_load_b128 = 1.0). Operands
// staged with global_load_async_to_lds_b128 into ping-pong LDS buffers;
// tile t+1 DMA issued before the WMMA block of tile t; s_wait_asynccnt 0
// + one barrier per k-tile. LDS row stride 72 bf16 (144B): 16B-aligned,
// conflict-free b128 fragment reads.
// =====================================================================
#define LSTR 72

template<bool RELU, bool C16>
__global__ __launch_bounds__(128)
void k_gemm(const unsigned short* __restrict__ A, int lda,
            const unsigned short* __restrict__ Bt, int ldbt,
            const float* __restrict__ bias,
            void* __restrict__ Cv, int ldc,
            int M, int N, int K)
{
  __shared__ unsigned short As[2][128 * LSTR];
  __shared__ unsigned short Bs[2][128 * LSTR];

  const int t    = threadIdx.x;
  const int lane = t & 31;
  const int wave = t >> 5;         // 0..3
  const int wm   = wave >> 1;      // 0..1
  const int wn   = wave & 1;       // 0..1
  const int row0 = blockIdx.y * 128;
  const int col0 = blockIdx.x * 128;
  const int ml   = lane & 15;
  const int hh   = lane >> 4;      // 0/1
  const int nk   = K >> 5;

  // staging: 1 thread per row, full 32-bf16 row = 64B = 4 async b128
  const unsigned long long abase =
      (unsigned long long)A + (size_t)(row0 + t) * (unsigned)lda * 2ull;
  const unsigned long long bbase =
      (unsigned long long)Bt + (size_t)(col0 + t) * (unsigned)ldbt * 2ull;

  f32x8 acc[4][4];
#pragma unroll
  for (int im = 0; im < 4; ++im)
#pragma unroll
    for (int in = 0; in < 4; ++in)
      acc[im][in] = (f32x8)0.0f;

  auto stage = [&](int kt, int buf) {
    const unsigned go = (unsigned)(kt * 32) * 2u;
    const unsigned la = lds_off(&As[buf][t * LSTR]);
    const unsigned lb = lds_off(&Bs[buf][t * LSTR]);
#pragma unroll
    for (int c = 0; c < 4; ++c) {
      async_cp_b128(la + c * 16u, go + c * 16u, abase);
      async_cp_b128(lb + c * 16u, go + c * 16u, bbase);
    }
  };

  // prologue: tile 0 -> buffer 0
  stage(0, 0);
  wait_async0();
  __syncthreads();

  for (int kt = 0; kt < nk; ++kt) {
    const int cur  = kt & 1;
    const int nxt  = cur ^ 1;
    const bool more = (kt + 1) < nk;

    if (more) stage(kt + 1, nxt);     // DMA for next tile runs under WMMA

    bf16x16 afr[4];
    bf16x16 bfr[4];
#pragma unroll
    for (int im = 0; im < 4; ++im) {
      const int r = wm * 64 + im * 16 + ml;
      union { u32x4 u[2]; bf16x16 v; } tmp;
      tmp.u[0] = *(const u32x4*)&As[cur][r * LSTR + hh * 8];        // K = h*8 .. +7
      tmp.u[1] = *(const u32x4*)&As[cur][r * LSTR + 16 + hh * 8];   // K = 16+h*8 .. +7
      afr[im] = tmp.v;
    }
#pragma unroll
    for (int in = 0; in < 4; ++in) {
      const int n = wn * 64 + in * 16 + ml;
      union { u32x4 u[2]; bf16x16 v; } tmp;
      const u32x4* p = (const u32x4*)&Bs[cur][n * LSTR + hh * 8];   // K = h*8 .. +7, 16+h*8 .. +7
      tmp.u[0] = p[0];
      tmp.u[1] = p[2];
      bfr[in] = tmp.v;
    }
#pragma unroll
    for (int im = 0; im < 4; ++im)
#pragma unroll
      for (int in = 0; in < 4; ++in)
        acc[im][in] = __builtin_amdgcn_wmma_f32_16x16x32_bf16(
            false, afr[im], false, bfr[in], (short)0, acc[im][in], false, false);

    if (more) wait_async0();
    __syncthreads();
  }

  // epilogue
#pragma unroll
  for (int im = 0; im < 4; ++im) {
#pragma unroll
    for (int in = 0; in < 4; ++in) {
      const int col = col0 + wn * 64 + in * 16 + ml;
      const float bv = bias ? bias[col] : 0.0f;
#pragma unroll
      for (int r = 0; r < 8; ++r) {
        const int row = row0 + wm * 64 + im * 16 + hh * 8 + r;
        float v = acc[im][in][r] + bv;
        if (RELU) v = fmaxf(v, 0.0f);
        if (C16) ((unsigned short*)Cv)[(size_t)row * ldc + col] = f2bf(v);
        else     ((float*)Cv)[(size_t)row * ldc + col] = v;
      }
    }
  }
}

// =====================================================================
// Fused attention per (b,h): scores = scale*Q*K^T (+mask), softmax, P*V.
// K,V cached in LDS as bf16; ctx written bf16 (feeds next GEMM's A).
// =====================================================================
template<int MAXKV, bool SELF>
__global__ __launch_bounds__(256)
void k_attn(const float* __restrict__ Q, const float* __restrict__ Km,
            const float* __restrict__ Vm, unsigned short* __restrict__ ctx,
            const int* __restrict__ caps, const int* __restrict__ order,
            int Lkv, int kvRowsPerB)
{
  __shared__ unsigned short K16[MAXKV * 66];
  __shared__ unsigned short V16[MAXKV * 66];
  __shared__ float qs[64];
  __shared__ float ss[256];
  __shared__ float red[256];
  __shared__ float pred[8 * 64];

  const int bh = blockIdx.x;
  const int b  = bh / HHH;
  const int h  = bh % HHH;
  const int t  = threadIdx.x;

  for (int idx = t; idx < Lkv * 64; idx += 256) {
    const int kv = idx >> 6, d = idx & 63;
    const size_t src = (size_t)(b * kvRowsPerB + kv) * (HHH * DKK) + h * DKK + d;
    K16[kv * 66 + d] = f2bf(Km[src]);
    V16[kv * 66 + d] = f2bf(Vm[src]);
  }
  int capk = 1;
  if (SELF && t < Lkv) capk = caps[order[b] * LL + t];
  __syncthreads();

  for (int q = 0; q < LL; ++q) {
    if (t < 64) qs[t] = Q[(size_t)(b * LL + q) * (HHH * DKK) + h * DKK + t];
    __syncthreads();

    float s;
    if (t < Lkv) {
      float acc = 0.0f;
#pragma unroll 8
      for (int e = 0; e < 64; e += 2) {
        const unsigned int w = *(const unsigned int*)&K16[t * 66 + e];
        acc += qs[e] * bf2f(w & 0xFFFFu) + qs[e + 1] * bf2f(w >> 16);
      }
      s = acc * 0.125f;                       // 1/sqrt(64)
      if (SELF) { if (t > q || capk == 0) s = -1.0e9f; }
    } else {
      s = -3.0e38f;
    }
    ss[t] = s;
    red[t] = s;
    __syncthreads();
    for (int st = 128; st > 0; st >>= 1) {
      if (t < st) red[t] = fmaxf(red[t], red[t + st]);
      __syncthreads();
    }
    const float mx = red[0];
    __syncthreads();

    const float p = (t < Lkv) ? __expf(ss[t] - mx) : 0.0f;
    red[t] = p;
    __syncthreads();
    for (int st = 128; st > 0; st >>= 1) {
      if (t < st) red[t] += red[t + st];
      __syncthreads();
    }
    const float inv = 1.0f / red[0];
    __syncthreads();
    ss[t] = p * inv;
    __syncthreads();

    const int d0 = (t & 31) * 2;
    const int part = t >> 5;
    float a0 = 0.0f, a1 = 0.0f;
    for (int kv = part; kv < Lkv; kv += 8) {
      const float pw = ss[kv];
      const unsigned int w = *(const unsigned int*)&V16[kv * 66 + d0];
      a0 += pw * bf2f(w & 0xFFFFu);
      a1 += pw * bf2f(w >> 16);
    }
    pred[part * 64 + d0]     = a0;
    pred[part * 64 + d0 + 1] = a1;
    __syncthreads();
    if (t < 64) {
      float o = 0.0f;
#pragma unroll
      for (int pp = 0; pp < 8; ++pp) o += pred[pp * 64 + t];
      ctx[(size_t)(b * LL + q) * (HHH * DKK) + h * DKK + t] = f2bf(o);
    }
    __syncthreads();
  }
}

// =====================================================================
// x = LayerNorm(tbuf + x) ; also writes bf16 mirror x16 (GEMM A operand)
// =====================================================================
__global__ __launch_bounds__(128)
void k_addln(const float* __restrict__ tbuf, float* __restrict__ x,
             unsigned short* __restrict__ x16)
{
  __shared__ float red[128];
  const int row = blockIdx.x, t = threadIdx.x;
  float v[4];
#pragma unroll
  for (int i = 0; i < 4; ++i) {
    const int d = t + i * 128;
    v[i] = tbuf[(size_t)row * DD + d] + x[(size_t)row * DD + d];
  }
  float s = v[0] + v[1] + v[2] + v[3];
  red[t] = s;
  __syncthreads();
  for (int st = 64; st > 0; st >>= 1) { if (t < st) red[t] += red[t + st]; __syncthreads(); }
  const float mean = red[0] * (1.0f / DD);
  __syncthreads();
  float sq = 0.0f;
#pragma unroll
  for (int i = 0; i < 4; ++i) { const float d = v[i] - mean; sq += d * d; }
  red[t] = sq;
  __syncthreads();
  for (int st = 64; st > 0; st >>= 1) { if (t < st) red[t] += red[t + st]; __syncthreads(); }
  const float rstd = rsqrtf(red[0] * (1.0f / DD) + 1e-5f);
#pragma unroll
  for (int i = 0; i < 4; ++i) {
    const float y = (v[i] - mean) * rstd;
    x[(size_t)row * DD + t + i * 128] = y;
    x16[(size_t)row * DD + t + i * 128] = f2bf(y);
  }
}

// stable descending argsort of 64 lengths
__global__ void k_sort(const int* __restrict__ lens, int* __restrict__ order)
{
  const int i = threadIdx.x;
  const int li = lens[i];
  int r = 0;
  for (int j = 0; j < BB; ++j) {
    const int lj = lens[j];
    r += (lj > li) || (lj == li && j < i);
  }
  order[r] = i;
}

// x[row] = tgt_emb[caps[order[b],l]] + pos_emb[l]  (+ bf16 mirror)
__global__ __launch_bounds__(256)
void k_embed(const int* __restrict__ caps, const int* __restrict__ order,
             const float* __restrict__ emb, const float* __restrict__ pos,
             float* __restrict__ x, unsigned short* __restrict__ x16)
{
  const int row = blockIdx.x;
  const int b = row / LL, l = row % LL;
  const int cap = caps[order[b] * LL + l];
  const int t = threadIdx.x;
#pragma unroll
  for (int i = 0; i < 2; ++i) {
    const int d = t + i * 256;
    const float v = emb[(size_t)cap * DD + d] + pos[l * DD + d];
    x[(size_t)row * DD + d] = v;
    x16[(size_t)row * DD + d] = f2bf(v);
  }
}

// enc_bf16[row] = bf16(encoder_out[order[b], p])   (row = b*196+p)
__global__ __launch_bounds__(256)
void k_gencbf(const float* __restrict__ enc, const int* __restrict__ order,
              unsigned short* __restrict__ out)
{
  const int row = blockIdx.x;
  const int b = row / NPIXX, p = row % NPIXX;
  const float* src = enc + ((size_t)(order[b] * NPIXX + p)) * ENCC;
  unsigned short* dst = out + (size_t)row * ENCC;
  for (int i = threadIdx.x; i < ENCC; i += 256) dst[i] = f2bf(src[i]);
}

// weight transpose+convert: out[N,K] bf16 = in[K,N] fp32 (32x32 LDS tiles)
__global__ __launch_bounds__(256)
void k_wt(const float* __restrict__ in, unsigned short* __restrict__ out,
          int Kd, int Nd)
{
  __shared__ float tile[32][33];
  const int k0 = blockIdx.x * 32, n0 = blockIdx.y * 32;
  const int tc = threadIdx.x & 31, tr = threadIdx.x >> 5;   // 32 cols x 8 rows
#pragma unroll
  for (int i = 0; i < 4; ++i) {
    const int k = tr + i * 8;
    tile[k][tc] = in[(size_t)(k0 + k) * Nd + n0 + tc];
  }
  __syncthreads();
#pragma unroll
  for (int i = 0; i < 4; ++i) {
    const int n = tr + i * 8;
    out[(size_t)(n0 + n) * Kd + k0 + tc] = f2bf(tile[tc][n]);
  }
}

// plain fp32 -> bf16 convert (count multiple of 2048)
__global__ __launch_bounds__(256)
void k_cvt(const float* __restrict__ in, unsigned short* __restrict__ out)
{
  int i = blockIdx.x * 2048 + threadIdx.x;
#pragma unroll
  for (int j = 0; j < 8; ++j, i += 256) out[i] = f2bf(in[i]);
}

// ---------------- host-side dispatch ----------------
static void launch_gemm(hipStream_t st, const unsigned short* A, int lda,
                        const unsigned short* Bt, int ldbt, const float* bias,
                        void* C, int ldc, int M, int N, int K,
                        bool relu_c16)
{
  dim3 g(N / 128, M / 128), blk(128);
  if (relu_c16) k_gemm<true,  true ><<<g, blk, 0, st>>>(A, lda, Bt, ldbt, bias, C, ldc, M, N, K);
  else          k_gemm<false, false><<<g, blk, 0, st>>>(A, lda, Bt, ldbt, bias, C, ldc, M, N, K);
}

extern "C" void kernel_launch(void* const* d_in, const int* in_sizes, int n_in,
                              void* d_out, int out_size, void* d_ws, size_t ws_size,
                              hipStream_t stream)
{
  const float* encoder_out = (const float*)d_in[0];
  const int*   caps        = (const int*)d_in[1];
  const int*   clen        = (const int*)d_in[2];
  const float* tgt_emb     = (const float*)d_in[3];
  const float* pos_emb     = (const float*)d_in[4];
  const float* sWq = (const float*)d_in[5];  const float* sbq = (const float*)d_in[6];
  const float* sWk = (const float*)d_in[7];  const float* sbk = (const float*)d_in[8];
  const float* sWv = (const float*)d_in[9];  const float* sbv = (const float*)d_in[10];
  const float* sWo = (const float*)d_in[11]; const float* sbo = (const float*)d_in[12];
  const float* eWq = (const float*)d_in[13]; const float* ebq = (const float*)d_in[14];
  const float* eWk = (const float*)d_in[15]; const float* ebk = (const float*)d_in[16];
  const float* eWv = (const float*)d_in[17]; const float* ebv = (const float*)d_in[18];
  const float* eWo = (const float*)d_in[19]; const float* ebo = (const float*)d_in[20];
  const float* fW1 = (const float*)d_in[21]; const float* fb1 = (const float*)d_in[22];
  const float* fW2 = (const float*)d_in[23]; const float* fb2 = (const float*)d_in[24];
  const float* projW = (const float*)d_in[25];
  float* out = (float*)d_out;

  typedef unsigned short ush;

  // scratch carve-out
  char* w = (char*)d_ws;
  auto carve = [&](size_t bytes) -> char* {
    char* p = w;
    w += (bytes + 255) & ~(size_t)255;
    return p;
  };
  int*   order = (int*)carve(BB * sizeof(int));
  float* X     = (float*)carve((size_t)MTOK * DD * 4);
  ush*   X16   = (ush*)carve((size_t)MTOK * DD * 2);
  float* T     = (float*)carve((size_t)MTOK * DD * 4);
  float* Qb    = (float*)carve((size_t)MTOK * DD * 4);
  float* Kb    = (float*)carve((size_t)MENC * DD * 4);
  float* Vb    = (float*)carve((size_t)MENC * DD * 4);
  ush*   Ctx16 = (ush*)carve((size_t)MTOK * DD * 2);
  ush*   Hb16  = (ush*)carve((size_t)MTOK * FFF * 2);
  ush*   EncBF = (ush*)carve((size_t)MENC * ENCC * 2);
  // pre-transposed bf16 weights [N,K] per layer
  ush* TsWq = (ush*)carve((size_t)NLAYER * DD * DD * 2);
  ush* TsWk = (ush*)carve((size_t)NLAYER * DD * DD * 2);
  ush* TsWv = (ush*)carve((size_t)NLAYER * DD * DD * 2);
  ush* TsWo = (ush*)carve((size_t)NLAYER * DD * DD * 2);
  ush* TeWq = (ush*)carve((size_t)NLAYER * DD * DD * 2);
  ush* TeWk = (ush*)carve((size_t)NLAYER * DD * ENCC * 2);
  ush* TeWv = (ush*)carve((size_t)NLAYER * DD * ENCC * 2);
  ush* TeWo = (ush*)carve((size_t)NLAYER * DD * DD * 2);
  ush* TfW1 = (ush*)carve((size_t)NLAYER * FFF * DD * 2);
  ush* TfW2 = (ush*)carve((size_t)NLAYER * DD * FFF * 2);
  ush* TprojW = (ush*)carve((size_t)VV * DD * 2);

  k_sort  <<<1, BB, 0, stream>>>(clen, order);
  k_embed <<<MTOK, 256, 0, stream>>>(caps, order, tgt_emb, pos_emb, X, X16);
  k_gencbf<<<MENC, 256, 0, stream>>>(encoder_out, order, EncBF);

  // one-time weight transpose+convert (in: [K,N] fp32 -> out: [N,K] bf16)
  for (int l = 0; l < NLAYER; ++l) {
    const dim3 gDD(DD / 32, DD / 32), gED(ENCC / 32, DD / 32),
               gDF(DD / 32, FFF / 32), gFD(FFF / 32, DD / 32), blk(256);
    k_wt<<<gDD, blk, 0, stream>>>(sWq + (size_t)l * DD * DD,  TsWq + (size_t)l * DD * DD,  DD, DD);
    k_wt<<<gDD, blk, 0, stream>>>(sWk + (size_t)l * DD * DD,  TsWk + (size_t)l * DD * DD,  DD, DD);
    k_wt<<<gDD, blk, 0, stream>>>(sWv + (size_t)l * DD * DD,  TsWv + (size_t)l * DD * DD,  DD, DD);
    k_wt<<<gDD, blk, 0, stream>>>(sWo + (size_t)l * DD * DD,  TsWo + (size_t)l * DD * DD,  DD, DD);
    k_wt<<<gDD, blk, 0, stream>>>(eWq + (size_t)l * DD * DD,  TeWq + (size_t)l * DD * DD,  DD, DD);
    k_wt<<<gED, blk, 0, stream>>>(eWk + (size_t)l * ENCC * DD, TeWk + (size_t)l * DD * ENCC, ENCC, DD);
    k_wt<<<gED, blk, 0, stream>>>(eWv + (size_t)l * ENCC * DD, TeWv + (size_t)l * DD * ENCC, ENCC, DD);
    k_wt<<<gDD, blk, 0, stream>>>(eWo + (size_t)l * DD * DD,  TeWo + (size_t)l * DD * DD,  DD, DD);
    k_wt<<<gDF, blk, 0, stream>>>(fW1 + (size_t)l * DD * FFF, TfW1 + (size_t)l * FFF * DD, DD, FFF);
    k_wt<<<gFD, blk, 0, stream>>>(fW2 + (size_t)l * FFF * DD, TfW2 + (size_t)l * DD * FFF, FFF, DD);
  }
  k_cvt<<<(VV * DD) / 2048, 256, 0, stream>>>(projW, TprojW);   // already [N,K]

  for (int l = 0; l < NLAYER; ++l) {
    const size_t oDD = (size_t)l * DD * DD;
    const size_t oED = (size_t)l * DD * ENCC;
    const size_t oDF = (size_t)l * FFF * DD;
    const size_t bD  = (size_t)l * DD;
    const size_t bF  = (size_t)l * FFF;

    // ---- self attention ----
    launch_gemm(stream, X16, DD, TsWq + oDD, DD, sbq + bD, Qb, DD, MTOK, DD, DD, false);
    launch_gemm(stream, X16, DD, TsWk + oDD, DD, sbk + bD, Kb, DD, MTOK, DD, DD, false);
    launch_gemm(stream, X16, DD, TsWv + oDD, DD, sbv + bD, Vb, DD, MTOK, DD, DD, false);
    k_attn<LL, true><<<BB * HHH, 256, 0, stream>>>(Qb, Kb, Vb, Ctx16, caps, order, LL, LL);
    launch_gemm(stream, Ctx16, DD, TsWo + oDD, DD, sbo + bD, T, DD, MTOK, DD, DD, false);
    k_addln<<<MTOK, 128, 0, stream>>>(T, X, X16);

    // ---- cross attention ----
    launch_gemm(stream, X16, DD, TeWq + oDD, DD, ebq + bD, Qb, DD, MTOK, DD, DD, false);
    launch_gemm(stream, EncBF, ENCC, TeWk + oED, ENCC, ebk + bD, Kb, DD, MENC, DD, ENCC, false);
    launch_gemm(stream, EncBF, ENCC, TeWv + oED, ENCC, ebv + bD, Vb, DD, MENC, DD, ENCC, false);
    k_attn<NPIXX, false><<<BB * HHH, 256, 0, stream>>>(Qb, Kb, Vb, Ctx16, nullptr, nullptr, NPIXX, NPIXX);
    launch_gemm(stream, Ctx16, DD, TeWo + oDD, DD, ebo + bD, T, DD, MTOK, DD, DD, false);
    k_addln<<<MTOK, 128, 0, stream>>>(T, X, X16);

    // ---- FFN ----
    launch_gemm(stream, X16, DD, TfW1 + oDF, DD, fb1 + bF, Hb16, FFF, MTOK, FFF, DD, true);
    launch_gemm(stream, Hb16, FFF, TfW2 + oDF, FFF, fb2 + bD, T, DD, MTOK, DD, FFF, false);
    k_addln<<<MTOK, 128, 0, stream>>>(T, X, X16);
  }

  // ---- final vocab projection: out = X @ projW^T ----
  launch_gemm(stream, X16, DD, TprojW, DD, nullptr, out, VV, MTOK, VV, DD, false);
}
